// MoE_46505905881795
// MI455X (gfx1250) — matrix operand order repaired and checked by reading, not compiled, not run on hardware
//
#include <hip/hip_runtime.h>

typedef __attribute__((ext_vector_type(16))) _Float16 v16h;
typedef __attribute__((ext_vector_type(8)))  _Float16 v8h;
typedef __attribute__((ext_vector_type(4)))  _Float16 v4h;
typedef __attribute__((ext_vector_type(2)))  _Float16 v2h;
typedef __attribute__((ext_vector_type(8)))  float    v8f;

#define B_TOK 8192
#define DDIM  1024
#define NEXP  8
#define TM    128
#define TN    128
#define TK    32
#define MOE_EPS 2.220446049250313e-16f

// ---------------------------------------------------------------------------
// Kernel 0: zero the output accumulator and the per-expert counters
// ---------------------------------------------------------------------------
__global__ void k_zero(float* __restrict__ out, int* __restrict__ counts, int n4) {
    int i = blockIdx.x * blockDim.x + threadIdx.x;
    if (i < n4) {
        float4 z = make_float4(0.f, 0.f, 0.f, 0.f);
        ((float4*)out)[i] = z;
    }
    if (blockIdx.x == 0 && threadIdx.x < NEXP) counts[threadIdx.x] = 0;
}

// ---------------------------------------------------------------------------
// Kernel 1: noisy top-2 gating + scatter tokens into per-expert lists
// ---------------------------------------------------------------------------
__global__ void k_gate(const float* __restrict__ x, const float* __restrict__ noise,
                       const float* __restrict__ wg, const float* __restrict__ wn,
                       int* __restrict__ counts, int* __restrict__ toks,
                       float* __restrict__ gvs) {
    __shared__ float dots[16][16];
    const int tl = threadIdx.x >> 4;
    const int j  = threadIdx.x & 15;
    const int token = blockIdx.x * 16 + tl;

    const float* xp = x + (size_t)token * DDIM;
    const float* wp = (j < 8) ? (wg + j) : (wn + (j - 8));

    float acc = 0.f;
#pragma unroll 4
    for (int d = 0; d < DDIM; d += 4) {
        float4 xv = *(const float4*)(xp + d);
        acc = fmaf(xv.x, wp[(d + 0) * NEXP], acc);
        acc = fmaf(xv.y, wp[(d + 1) * NEXP], acc);
        acc = fmaf(xv.z, wp[(d + 2) * NEXP], acc);
        acc = fmaf(xv.w, wp[(d + 3) * NEXP], acc);
    }
    dots[tl][j] = acc;
    __syncthreads();

    if (j == 0) {
        float lg[NEXP];
#pragma unroll
        for (int e = 0; e < NEXP; ++e) {
            float clean = dots[tl][e];
            float v     = dots[tl][e + 8];
            float sp    = log1pf(expf(v)) + 0.01f;        // softplus + 0.01
            lg[e] = clean + noise[(size_t)token * NEXP + e] * sp;
        }
        int i1 = 0;
#pragma unroll
        for (int e = 1; e < NEXP; ++e) if (lg[e] > lg[i1]) i1 = e;
        int i2 = (i1 == 0) ? 1 : 0;
#pragma unroll
        for (int e = 0; e < NEXP; ++e) if (e != i1 && lg[e] > lg[i2]) i2 = e;

        float m  = lg[i1];
        float e2 = expf(lg[i2] - m);
        float s  = 1.f + e2;

        int s1 = atomicAdd(&counts[i1], 1);
        toks[i1 * B_TOK + s1] = token;  gvs[i1 * B_TOK + s1] = 1.f / s;
        int s2 = atomicAdd(&counts[i2], 1);
        toks[i2 * B_TOK + s2] = token;  gvs[i2 * B_TOK + s2] = e2 / s;
    }
}

// ---------------------------------------------------------------------------
// v8h pair -> v16h fragment (16 contiguous halfs = two 16B LDS loads)
// ---------------------------------------------------------------------------
__device__ __forceinline__ v16h load_frag(const _Float16* base) {
    v8h lo = *(const v8h*)(base);
    v8h hi = *(const v8h*)(base + 8);
    return __builtin_shufflevector(lo, hi, 0, 1, 2, 3, 4, 5, 6, 7,
                                           8, 9, 10, 11, 12, 13, 14, 15);
}

// ---------------------------------------------------------------------------
// Kernel 2: routed expert GEMM, fused bias + exp + gate scale + atomic combine
//   grid = (64 m-tile slots, 8 n-tiles, 8 experts); extra slots exit early.
//   128x128 tile, K-step 32, double-buffered LDS, f32->f16 convert in staging.
//   8 waves -> 4 (M) x 2 (N); each wave owns 2x4 16x16 WMMA subtiles.
//   All tile base pointers hoisted out of the K loop; no divergence in staging.
// ---------------------------------------------------------------------------
__global__ __launch_bounds__(256, 1)
void k_expert(const float* __restrict__ x, const float* __restrict__ We,
              const float* __restrict__ be, const int* __restrict__ counts,
              const int* __restrict__ toks, const float* __restrict__ gvs,
              float* __restrict__ out) {
    const int e   = blockIdx.z;
    const int cnt = counts[e];
    const int mt  = blockIdx.x;
    if (mt * TM >= cnt) return;
    const int nt   = blockIdx.y;
    const int rows = (cnt - mt * TM < TM) ? (cnt - mt * TM) : TM;

    __shared__ alignas(16) _Float16 Alds[2][TM * TK];   // [m*32 + k]
    __shared__ alignas(16) _Float16 Blds[2][TN * TK];   // [n*32 + k] (K-contig per col)
    __shared__ int   stok[TM];
    __shared__ float sgv[TM];

    const int tid = threadIdx.x;
    if (tid < TM) {
        if (tid < rows) {
            stok[tid] = toks[e * B_TOK + mt * TM + tid];
            sgv[tid]  = gvs [e * B_TOK + mt * TM + tid];
        } else {
            stok[tid] = -1;            // masked in epilogue
            sgv[tid]  = 0.f;
        }
    }
    __syncthreads();

    const float* Wbase = We + (size_t)e * DDIM * DDIM + (size_t)nt * TN;

    const int wid  = tid >> 5;
    const int lane = tid & 31;
    const int wm   = wid & 3;                // 4 waves along M (32 rows each)
    const int wn   = wid >> 2;               // 2 waves along N (64 cols each)
    const int lrow = lane & 15;
    const int kA   = (lane < 16) ? 0 : 16;   // A: lanes 16-31 hold K=16..31
    const int kB   = (lane < 16) ? 0 : 16;   // B: lanes 16-31 hold K=16..31

    // ---- hoist all staging base pointers out of the K loop (loop-invariant) ----
    const float* pa[4];      // A: 4 slots, each = one float4 of one gathered row
    int am[4], aq[4];
#pragma unroll
    for (int i = 0; i < 4; ++i) {
        int idx = i * 256 + tid;
        am[i] = idx >> 3;                    // row in tile  (0..127)
        aq[i] = idx & 7;                     // k-quad       (0..7)
        int token = stok[am[i]];
        if (token < 0) token = 0;            // clamp: no divergence; masked later
        pa[i] = x + (size_t)token * DDIM + aq[i] * 4;
    }
    const float* pb[2];      // B: 2 slots, each = k-pair x one n-quad
    int bkp[2], bq[2];
#pragma unroll
    for (int i = 0; i < 2; ++i) {
        int idx = i * 256 + tid;
        bkp[i] = idx >> 5;                   // k-pair (0..15) -> rows 2kp, 2kp+1
        bq[i]  = idx & 31;                   // n-quad (0..31)
        pb[i] = Wbase + (size_t)(2 * bkp[i]) * DDIM + bq[i] * 4;
    }

    float4 ra[4], rb0[2], rb1[2];
    auto loadRegs = [&](int kt) {
#pragma unroll
        for (int i = 0; i < 4; ++i)
            ra[i] = *(const float4*)(pa[i] + kt * TK);
#pragma unroll
        for (int i = 0; i < 2; ++i) {
            const float* p = pb[i] + (size_t)kt * TK * DDIM;
            rb0[i] = *(const float4*)(p);
            rb1[i] = *(const float4*)(p + DDIM);
        }
    };
    auto storeLds = [&](int buf) {
#pragma unroll
        for (int i = 0; i < 4; ++i) {
            v4h h = { (_Float16)ra[i].x, (_Float16)ra[i].y,
                      (_Float16)ra[i].z, (_Float16)ra[i].w };
            *(v4h*)&Alds[buf][am[i] * TK + aq[i] * 4] = h;
        }
#pragma unroll
        for (int i = 0; i < 2; ++i) {        // transpose to [n][k], K-pairs packed
            int kcol = 2 * bkp[i];
            int n0   = bq[i] * 4;
            v2h p0 = { (_Float16)rb0[i].x, (_Float16)rb1[i].x };
            v2h p1 = { (_Float16)rb0[i].y, (_Float16)rb1[i].y };
            v2h p2 = { (_Float16)rb0[i].z, (_Float16)rb1[i].z };
            v2h p3 = { (_Float16)rb0[i].w, (_Float16)rb1[i].w };
            *(v2h*)&Blds[buf][(n0 + 0) * TK + kcol] = p0;
            *(v2h*)&Blds[buf][(n0 + 1) * TK + kcol] = p1;
            *(v2h*)&Blds[buf][(n0 + 2) * TK + kcol] = p2;
            *(v2h*)&Blds[buf][(n0 + 3) * TK + kcol] = p3;
        }
    };

    v8f acc[2][4];
#pragma unroll
    for (int ms = 0; ms < 2; ++ms)
#pragma unroll
        for (int ns = 0; ns < 4; ++ns)
            acc[ms][ns] = (v8f){0.f, 0.f, 0.f, 0.f, 0.f, 0.f, 0.f, 0.f};

    const int NK = DDIM / TK;                // 32 K-steps
    loadRegs(0);
    storeLds(0);
    __syncthreads();

    for (int kt = 0; kt < NK; ++kt) {
        const int cur   = kt & 1;
        const bool more = (kt + 1 < NK);
        if (more) loadRegs(kt + 1);
        if (kt + 2 < NK)                     // warm GL2 for the K-step after next
            __builtin_prefetch(Wbase + (size_t)((kt + 2) * TK + (tid & 31)) * DDIM, 0, 1);

        v16h af[2], bf[4];
#pragma unroll
        for (int ms = 0; ms < 2; ++ms)
            af[ms] = load_frag(&Alds[cur][(wm * 32 + ms * 16 + lrow) * TK + kA]);
#pragma unroll
        for (int ns = 0; ns < 4; ++ns)
            bf[ns] = load_frag(&Blds[cur][(wn * 64 + ns * 16 + lrow) * TK + kB]);

#pragma unroll
        for (int ms = 0; ms < 2; ++ms)
#pragma unroll
            for (int ns = 0; ns < 4; ++ns)
                acc[ms][ns] = __builtin_amdgcn_wmma_f32_16x16x32_f16(
                    false, af[ms], false, bf[ns], (short)0, acc[ms][ns], false, false);

        if (more) storeLds(cur ^ 1);
        __syncthreads();
    }

    // Epilogue: D[m][n] -> gate * exp(D + bias)  accumulated atomically per token
#pragma unroll
    for (int ms = 0; ms < 2; ++ms) {
#pragma unroll
        for (int ns = 0; ns < 4; ++ns) {
            const int ncol = nt * TN + wn * 64 + ns * 16 + (lane & 15);
            const float bias = be[(size_t)e * DDIM + ncol];
#pragma unroll
            for (int r = 0; r < 8; ++r) {
                int mrow = wm * 32 + ms * 16 + r + ((lane < 16) ? 0 : 8);
                if (mrow < rows) {
                    int token = stok[mrow];
                    float v = acc[ms][ns][r] + bias;
                    atomicAdd(out + (size_t)token * DDIM + ncol, sgv[mrow] * expf(v));
                }
            }
        }
    }
}

// ---------------------------------------------------------------------------
// Kernel 3: in-place log(combined), exact zeros replaced by EPS
// ---------------------------------------------------------------------------
__global__ void k_log(float* __restrict__ out, int n) {
    int i = blockIdx.x * blockDim.x + threadIdx.x;
    if (i < n) {
        float v = out[i];
        out[i] = logf(v == 0.f ? MOE_EPS : v);
    }
}

// ---------------------------------------------------------------------------
extern "C" void kernel_launch(void* const* d_in, const int* in_sizes, int n_in,
                              void* d_out, int out_size, void* d_ws, size_t ws_size,
                              hipStream_t stream) {
    const float* x     = (const float*)d_in[0];
    const float* noise = (const float*)d_in[1];
    const float* wg    = (const float*)d_in[2];
    const float* wn    = (const float*)d_in[3];
    const float* We    = (const float*)d_in[4];
    const float* be    = (const float*)d_in[5];
    float* out = (float*)d_out;

    char* ws      = (char*)d_ws;
    int*   counts = (int*)ws;                                        // 8 ints
    int*   toks   = (int*)(ws + 256);                                // 8 x 8192 ints
    float* gvs    = (float*)(ws + 256 + NEXP * B_TOK * sizeof(int)); // 8 x 8192 floats

    const int n4 = (B_TOK * DDIM) / 4;
    k_zero<<<dim3((n4 + 255) / 256), dim3(256), 0, stream>>>(out, counts, n4);

    k_gate<<<dim3(B_TOK / 16), dim3(256), 0, stream>>>(x, noise, wg, wn,
                                                       counts, toks, gvs);

    k_expert<<<dim3(B_TOK / TM, DDIM / TN, NEXP), dim3(256), 0, stream>>>(
        x, We, be, counts, toks, gvs, out);

    const int n = B_TOK * DDIM;
    k_log<<<dim3((n + 255) / 256), dim3(256), 0, stream>>>(out, n);
}
